// LSTM_84782654423271
// MI455X (gfx1250) — matrix operand order repaired and checked by reading, not compiled, not run on hardware
//
#include <hip/hip_runtime.h>
#include <hip/hip_bf16.h>

// LSTM: B=64, T=256, D=1024, F=1024
// Phase 1: xz = bf16(x @ Wi + b)  — one big WMMA GEMM, fully parallel.
// Phase 2: ONE persistent kernel runs all 256 recurrent steps:
//   - Wh fragments live in VGPRs for the entire scan (128 VGPRs/wave)
//   - c lives in wave-0 registers for the entire scan
//   - steps separated by a software grid barrier (atomic counter + fences)
//   - h double-buffered in global (L2-resident, 128 KB)

typedef __attribute__((ext_vector_type(16))) __bf16 v16bf;
typedef __attribute__((ext_vector_type(8)))  __bf16 v8bf;
typedef __attribute__((ext_vector_type(8)))  float  v8f;
typedef __attribute__((ext_vector_type(4)))  float  v4f;

#define BB 64
#define TT 256
#define DD 1024
#define FF 1024
#define G4 4096

__device__ __forceinline__ unsigned short f32_to_bf16(float f) {
  unsigned int u = __float_as_uint(f);
  u += 0x7FFFu + ((u >> 16) & 1u);          // round-to-nearest-even
  return (unsigned short)(u >> 16);
}
__device__ __forceinline__ float bf16_to_f32(unsigned short h) {
  return __uint_as_float(((unsigned int)h) << 16);
}
__device__ __forceinline__ float sigmoid_f(float x) {
  return 1.0f / (1.0f + __expf(-x));
}
__device__ __forceinline__ float tanh_f(float x) {
  x = fminf(fmaxf(x, -15.0f), 15.0f);
  float e = __expf(2.0f * x);
  return (e - 1.0f) / (e + 1.0f);
}
__device__ __forceinline__ v16bf make_a(v8bf lo, v8bf hi) {
  v16bf a;
#pragma unroll
  for (int j = 0; j < 8; ++j) { a[j] = lo[j]; a[j + 8] = hi[j]; }
  return a;
}
#define WMMA_BF16(A, Bm, C) \
  __builtin_amdgcn_wmma_f32_16x16x32_bf16(false, (A), false, (Bm), (short)0, (C), false, false)

// ---------------- conversion / init kernels ----------------

__global__ void cvt_f32_bf16(const float* __restrict__ src,
                             unsigned short* __restrict__ dst, int n) {
  int i = (blockIdx.x * blockDim.x + threadIdx.x) * 4;
  if (i + 3 < n) {
    v4f v = *(const v4f*)(src + i);
#pragma unroll
    for (int j = 0; j < 4; ++j) dst[i + j] = f32_to_bf16(v[j]);
  }
}

// dst[c*R + r] = bf16(src[r*C + c])   (R x C row-major -> transposed bf16)
__global__ void transpose_to_bf16(const float* __restrict__ src,
                                  unsigned short* __restrict__ dst,
                                  int R, int C) {
  __shared__ float tile[32][33];
  int c0 = blockIdx.x * 32, r0 = blockIdx.y * 32;
  int tx = threadIdx.x, ty = threadIdx.y;   // 32 x 8
#pragma unroll
  for (int i = 0; i < 32; i += 8)
    tile[ty + i][tx] = src[(size_t)(r0 + ty + i) * C + (c0 + tx)];
  __syncthreads();
#pragma unroll
  for (int i = 0; i < 32; i += 8)
    dst[(size_t)(c0 + ty + i) * R + (r0 + tx)] = f32_to_bf16(tile[tx][ty + i]);
}

__global__ void init_state(const float* __restrict__ h0,
                           const float* __restrict__ c0,
                           unsigned short* __restrict__ hb,
                           float* __restrict__ cbuf,
                           unsigned int* __restrict__ cnt, int n) {
  int i = blockIdx.x * blockDim.x + threadIdx.x;
  if (i < n) { hb[i] = f32_to_bf16(h0[i]); cbuf[i] = c0[i]; }
  if (i == 0) *cnt = 0u;                    // reset grid barrier each call
}

// ---------------- phase 1: xz = x @ Wi + b ----------------
__global__ __launch_bounds__(128)
void xz_gemm(const unsigned short* __restrict__ xb,
             const unsigned short* __restrict__ WiT,
             const float* __restrict__ bias,
             unsigned short* __restrict__ xz) {
  const int lane = threadIdx.x & 31;
  const int wave = threadIdx.x >> 5;
  const int kg   = lane >> 4;
  const int l16  = lane & 15;
  const int m0 = blockIdx.y * 16;
  const int n0 = blockIdx.x * 256 + wave * 64;

  const unsigned short* aptr = xb  + (size_t)(m0 + l16) * DD + 16 * kg;
  const unsigned short* bptr = WiT + (size_t)(n0 + l16) * DD + 16 * kg;
  const size_t GS = (size_t)16 * DD;

  v8f acc0 = {}, acc1 = {}, acc2 = {}, acc3 = {};
  for (int k0 = 0; k0 < DD; k0 += 32) {
    v16bf A = make_a(*(const v8bf*)(aptr + k0), *(const v8bf*)(aptr + k0 + 8));
    v16bf B0 = *(const v16bf*)(bptr + k0);
    v16bf B1 = *(const v16bf*)(bptr + GS + k0);
    v16bf B2 = *(const v16bf*)(bptr + 2 * GS + k0);
    v16bf B3 = *(const v16bf*)(bptr + 3 * GS + k0);
    acc0 = WMMA_BF16(A, B0, acc0);
    acc1 = WMMA_BF16(A, B1, acc1);
    acc2 = WMMA_BF16(A, B2, acc2);
    acc3 = WMMA_BF16(A, B3, acc3);
  }
  v8f accs[4] = {acc0, acc1, acc2, acc3};
#pragma unroll
  for (int g = 0; g < 4; ++g) {
    int n = n0 + 16 * g + l16;
    float bn = bias[n];
#pragma unroll
    for (int v = 0; v < 8; ++v) {
      int row = m0 + v + 8 * kg;
      xz[(size_t)row * G4 + n] = f32_to_bf16(accs[g][v] + bn);
    }
  }
}

// ---------------- phase 2: persistent scan kernel ----------------
// grid (F/16=64, B/32=2) = 128 blocks, 256 threads = 8 waves.
// Wave w holds Wh[k in w*128..w*128+128) for its 16 cols x 4 gates in VGPRs
// (4 k-iters x 4 gates x v16bf = 128 VGPRs), resident across all 256 steps.
// Per step: load h fragments, 32 WMMAs, LDS split-K reduction, wave 0 applies
// gates + updates register-resident c, writes out + bf16 h; grid barrier.
__global__ __launch_bounds__(256)
void lstm_scan(unsigned short* __restrict__ hb0,
               unsigned short* __restrict__ hb1,
               const unsigned short* __restrict__ WhT,   // (4F,F) bf16
               const unsigned short* __restrict__ xz,    // (B*T,4F) bf16
               const float* __restrict__ cinit,          // (B,F) f32
               float* __restrict__ out,                  // (B,T,F) f32
               unsigned int* __restrict__ cnt) {
  __shared__ __align__(32) float red[7][8][32][8];   // 56 KB

  const int lane = threadIdx.x & 31;
  const int wave = threadIdx.x >> 5;       // 0..7
  const int kg  = lane >> 4;
  const int l16 = lane & 15;
  const int f0 = blockIdx.x * 16;
  const int m0 = blockIdx.y * 32;          // 2 row-tiles: m0, m0+16
  const size_t GS = (size_t)FF * FF;       // gate stride in WhT
  const int kbeg = wave * 128;             // split-K over 8 waves
  const unsigned int NB = gridDim.x * gridDim.y;
  const int col = f0 + l16;

  // ---- Wh fragments resident in VGPRs for the whole scan ----
  const unsigned short* bptr = WhT + (size_t)(f0 + l16) * FF + 16 * kg + kbeg;
  v16bf Bf[4][4];                          // [k-iter][gate]
#pragma unroll
  for (int i = 0; i < 4; ++i)
#pragma unroll
    for (int g = 0; g < 4; ++g)
      Bf[i][g] = *(const v16bf*)(bptr + (size_t)g * GS + i * 32);

  // ---- c resident in wave-0 registers for the whole scan ----
  float creg[16];                          // [tile*8 + v]
  if (wave == 0) {
#pragma unroll
    for (int tile = 0; tile < 2; ++tile)
#pragma unroll
      for (int v = 0; v < 8; ++v)
        creg[tile * 8 + v] = cinit[(m0 + tile * 16 + v + 8 * kg) * FF + col];
  }

  for (int t = 0; t < TT; ++t) {
    const unsigned short* hin  = (t & 1) ? hb1 : hb0;
    unsigned short*       hout = (t & 1) ? hb0 : hb1;

    const unsigned short* aptr0 = hin + (m0 + l16) * FF + 16 * kg + kbeg;
    const unsigned short* aptr1 = aptr0 + 16 * FF;

    v8f acc[8] = {};                       // [0..3]=tile0 gates, [4..7]=tile1
#pragma unroll
    for (int i = 0; i < 4; ++i) {
      v16bf A0 = make_a(*(const v8bf*)(aptr0 + i * 32),
                        *(const v8bf*)(aptr0 + i * 32 + 8));
      v16bf A1 = make_a(*(const v8bf*)(aptr1 + i * 32),
                        *(const v8bf*)(aptr1 + i * 32 + 8));
      acc[0] = WMMA_BF16(A0, Bf[i][0], acc[0]);
      acc[1] = WMMA_BF16(A0, Bf[i][1], acc[1]);
      acc[2] = WMMA_BF16(A0, Bf[i][2], acc[2]);
      acc[3] = WMMA_BF16(A0, Bf[i][3], acc[3]);
      acc[4] = WMMA_BF16(A1, Bf[i][0], acc[4]);
      acc[5] = WMMA_BF16(A1, Bf[i][1], acc[5]);
      acc[6] = WMMA_BF16(A1, Bf[i][2], acc[6]);
      acc[7] = WMMA_BF16(A1, Bf[i][3], acc[7]);
    }

    if (wave > 0) {
#pragma unroll
      for (int a = 0; a < 8; ++a)
        *(v8f*)&red[wave - 1][a][lane][0] = acc[a];
    }
    __syncthreads();

    if (wave == 0) {
#pragma unroll
      for (int w = 0; w < 7; ++w)
#pragma unroll
        for (int a = 0; a < 8; ++a)
          acc[a] += *(const v8f*)&red[w][a][lane][0];

#pragma unroll
      for (int tile = 0; tile < 2; ++tile) {
#pragma unroll
        for (int v = 0; v < 8; ++v) {
          int b = m0 + tile * 16 + v + 8 * kg;
          const unsigned short* xzp = xz + ((size_t)b * TT + t) * G4;
          float zi = acc[4 * tile + 0][v] + bf16_to_f32(xzp[col]);
          float zf = acc[4 * tile + 1][v] + bf16_to_f32(xzp[FF + col]);
          float zg = acc[4 * tile + 2][v] + bf16_to_f32(xzp[2 * FF + col]);
          float zo = acc[4 * tile + 3][v] + bf16_to_f32(xzp[3 * FF + col]);
          float ig = sigmoid_f(zi);
          float fg = sigmoid_f(zf);
          float gg = tanh_f(zg);
          float og = sigmoid_f(zo);
          float cn = fg * creg[tile * 8 + v] + ig * gg;
          float hn = og * tanh_f(cn);
          creg[tile * 8 + v] = cn;
          out[((size_t)b * TT + t) * FF + col] = hn;
          hout[b * FF + col] = f32_to_bf16(hn);
        }
      }
    }

    // ---- grid barrier: release stores, arrive, spin, acquire ----
    __threadfence();                       // release h/out stores to device
    __syncthreads();                       // all waves fenced
    if (threadIdx.x == 0) {
      __hip_atomic_fetch_add(cnt, 1u, __ATOMIC_RELEASE, __HIP_MEMORY_SCOPE_AGENT);
      unsigned int tgt = NB * (unsigned int)(t + 1);
      while (__hip_atomic_load(cnt, __ATOMIC_ACQUIRE, __HIP_MEMORY_SCOPE_AGENT) < tgt)
        __builtin_amdgcn_s_sleep(2);
    }
    __syncthreads();
    __threadfence();                       // conservative acquire for all waves
  }
}

// ---------------- launch ----------------

extern "C" void kernel_launch(void* const* d_in, const int* in_sizes, int n_in,
                              void* d_out, int out_size, void* d_ws, size_t ws_size,
                              hipStream_t stream) {
  const float* x  = (const float*)d_in[0];
  const float* h0 = (const float*)d_in[1];
  const float* c0 = (const float*)d_in[2];
  const float* Wi = (const float*)d_in[3];
  const float* Wh = (const float*)d_in[4];
  const float* bv = (const float*)d_in[5];
  float* out = (float*)d_out;

  char* ws = (char*)d_ws;
  size_t off = 0;
  unsigned short* xz  = (unsigned short*)(ws + off); off += (size_t)BB * TT * G4 * 2;  // 134 MB
  unsigned short* xb  = (unsigned short*)(ws + off); off += (size_t)BB * TT * DD * 2;  //  33 MB
  unsigned short* WiT = (unsigned short*)(ws + off); off += (size_t)G4 * DD * 2;       //   8 MB
  unsigned short* WhT = (unsigned short*)(ws + off); off += (size_t)G4 * FF * 2;       //   8 MB
  unsigned short* hb0 = (unsigned short*)(ws + off); off += (size_t)BB * FF * 2;
  unsigned short* hb1 = (unsigned short*)(ws + off); off += (size_t)BB * FF * 2;
  float*          cb  = (float*)(ws + off);          off += (size_t)BB * FF * 4;
  unsigned int*   cnt = (unsigned int*)(ws + off);   off += 256;
  (void)ws_size; (void)in_sizes; (void)n_in; (void)out_size;

  // x -> bf16
  {
    int n = BB * TT * DD;
    cvt_f32_bf16<<<n / 4 / 256, 256, 0, stream>>>(x, xb, n);
  }
  // Wi (D x 4F) -> WiT (4F x D) bf16 ; Wh (F x 4F) -> WhT (4F x F) bf16
  transpose_to_bf16<<<dim3(G4 / 32, DD / 32), dim3(32, 8), 0, stream>>>(Wi, WiT, DD, G4);
  transpose_to_bf16<<<dim3(G4 / 32, FF / 32), dim3(32, 8), 0, stream>>>(Wh, WhT, FF, G4);
  // h0/c0 -> state buffers, reset grid barrier
  init_state<<<(BB * FF) / 256, 256, 0, stream>>>(h0, c0, hb0, cb, cnt, BB * FF);

  // Phase 1: xz = x @ Wi + b
  xz_gemm<<<dim3(G4 / 256, (BB * TT) / 16), 128, 0, stream>>>(xb, WiT, bv, xz);

  // Phase 2: entire scan in one persistent kernel (software grid barrier)
  lstm_scan<<<dim3(FF / 16, BB / 32), 256, 0, stream>>>(hb0, hb1, WhT, xz, cb, out, cnt);
}
